// SwinTransformerBlock_61203283968862
// MI455X (gfx1250) — matrix operand order repaired and checked by reading, not compiled, not run on hardware
//
#include <hip/hip_runtime.h>
#include <hip/hip_bf16.h>
#include <math.h>

#define DIM   384
#define HEADS 12
#define NWIN  49
#define MLPH  1536
#define LTOK  3136
#define HH    56
#define MTOT  100352   // 32*3136 = 2048 windows * 49
#define NWTOT 2048
#define SCALE 0.17677669529663687f   // 32^-0.5

typedef unsigned short u16;
typedef unsigned int   u32;
typedef __attribute__((ext_vector_type(16))) __bf16 v16bf;
typedef __attribute__((ext_vector_type(8)))  float  v8f;

union BFV { v16bf v; u16 u[16]; uint4 q[2]; };

__device__ __forceinline__ u16 f2bf(float f) {
  union { float f; u32 u; } c; c.f = f;
  u32 u = c.u;
  return (u16)((u + 0x7FFFu + ((u >> 16) & 1u)) >> 16);   // RNE
}

__device__ __forceinline__ v8f vzero8() {
  v8f z;
#pragma unroll
  for (int i = 0; i < 8; ++i) z[i] = 0.f;
  return z;
}

__device__ __forceinline__ v8f wmma_bf16(v16bf a, v16bf b, v8f c) {
  return __builtin_amdgcn_wmma_f32_16x16x32_bf16(false, a, false, b, (short)0, c, false, false);
}

// A fragment (16x32 tile at row0/k0, LDS row-major, ld u16; (ld*2)%16==0):
// lane m=lane%16 holds A[m][kb..kb+8) and A[m][kb+16..kb+24), kb=0/8 by lane half.
__device__ __forceinline__ v16bf load_frag_a(const u16* lds, int ld, int row0, int k0, int lane) {
  const u16* p = lds + (row0 + (lane & 15)) * ld + k0 + ((lane < 16) ? 0 : 8);
  BFV r;
  r.q[0] = *(const uint4*)p;
  r.q[1] = *(const uint4*)(p + 16);
  return r.v;
}

// B fragment from TRANSPOSED staging Bt[n][k] (ld u16, (ld*2)%16==0):
// lane holds col n=n0+lane%16, K slices kb..kb+8 and kb+16..kb+24, kb=0/8 by lane half.
__device__ __forceinline__ v16bf load_frag_bt(const u16* lds, int ld, int n0, int k0, int lane) {
  const u16* p = lds + (n0 + (lane & 15)) * ld + k0 + ((lane < 16) ? 0 : 8);
  BFV r;
  r.q[0] = *(const uint4*)p;
  r.q[1] = *(const uint4*)(p + 16);
  return r.v;
}

// window-major row r -> token index
__device__ __forceinline__ int row_to_token(int r) {
  int wi = r / NWIN, n = r % NWIN;
  int b = wi >> 6, rem = wi & 63;
  int h = (rem >> 3) * 7 + n / 7;
  int w = (rem & 7) * 7 + n % 7;
  return b * LTOK + h * HH + w;
}

// ---------------- small prep kernels ----------------

// Wt[n][k] = bf16(W[k][n])  (transpose so GEMM B-tiles load coalesced and frag-contiguous)
__global__ void transpose_bf16_kernel(const float* __restrict__ W, u16* __restrict__ Wt,
                                      int K, int N) {
  int i = blockIdx.x * blockDim.x + threadIdx.x;
  if (i >= K * N) return;
  int n = i / K, k = i % K;
  Wt[i] = f2bf(W[(size_t)k * N + n]);
}

__global__ void bias_expand_kernel(const float* __restrict__ tab, const int* __restrict__ ridx,
                                   float* __restrict__ biasW) {
  int i = blockIdx.x * blockDim.x + threadIdx.x;
  if (i >= HEADS * NWIN * NWIN) return;
  int h = i / (NWIN * NWIN);
  int rem = i % (NWIN * NWIN);
  biasW[i] = tab[ridx[rem] * HEADS + h];
}

// LayerNorm over 384; one wave per row. permuted=1: read token t=perm(r), write row r.
__global__ __launch_bounds__(256) void ln_kernel(const float* __restrict__ xin,
                                                 const float* __restrict__ g,
                                                 const float* __restrict__ b,
                                                 u16* __restrict__ out, int permuted) {
  int wave = threadIdx.x >> 5, lane = threadIdx.x & 31;
  int r = blockIdx.x * 8 + wave;
  int t = permuted ? row_to_token(r) : r;
  const float* xp = xin + (size_t)t * DIM;
  float v[12], s = 0.f, s2 = 0.f;
#pragma unroll
  for (int j = 0; j < 12; ++j) { float f = xp[lane + 32 * j]; v[j] = f; s += f; s2 += f * f; }
#pragma unroll
  for (int m = 1; m < 32; m <<= 1) { s += __shfl_xor(s, m, 32); s2 += __shfl_xor(s2, m, 32); }
  float mean = s * (1.f / DIM);
  float var  = s2 * (1.f / DIM) - mean * mean;
  float rs   = rsqrtf(var + 1e-5f);
  u16* op = out + (size_t)r * DIM;
#pragma unroll
  for (int j = 0; j < 12; ++j) {
    int c = lane + 32 * j;
    op[c] = f2bf((v[j] - mean) * rs * g[c] + b[c]);
  }
}

// ---------------- 128x128 GEMM core ----------------
// 256 threads / 8 waves; wave = 32 rows x 64 cols = 2x4 fragments; BK=32.
// A row-major (ldA); Bt transposed weights [N][K]. Register double-buffered tiles.
__device__ __forceinline__ void gemm_core(const u16* __restrict__ A, const u16* __restrict__ Bt,
                                          int K, int ldA, int rowBase, int colBase,
                                          u16* As, u16* Bs, v8f acc[2][4]) {
  const int tid = threadIdx.x;
  const int lane = tid & 31;
  const int wave = tid >> 5;
  const int rowOfs = (wave >> 1) * 32;
  const int colOfs = (wave & 1) * 64;
  const int ar = tid >> 1, ac = (tid & 1) * 16;     // A: 128 rows x 32 k
  const int bn = tid >> 1, bc = (tid & 1) * 16;     // Bt: 128 n  x 32 k
  const u16* aSrc = A + (size_t)(rowBase + ar) * ldA;
  const u16* bSrc = Bt + (size_t)(colBase + bn) * K;

  uint4 ra0 = *(const uint4*)(aSrc + ac);
  uint4 ra1 = *(const uint4*)(aSrc + ac + 8);
  uint4 rb0 = *(const uint4*)(bSrc + bc);
  uint4 rb1 = *(const uint4*)(bSrc + bc + 8);

  for (int k0 = 0; k0 < K; k0 += 32) {
    __syncthreads();
    *(uint4*)(As + ar * 48 + ac)     = ra0;
    *(uint4*)(As + ar * 48 + ac + 8) = ra1;
    *(uint4*)(Bs + bn * 40 + bc)     = rb0;
    *(uint4*)(Bs + bn * 40 + bc + 8) = rb1;
    __syncthreads();
    if (k0 + 32 < K) {
      ra0 = *(const uint4*)(aSrc + k0 + 32 + ac);
      ra1 = *(const uint4*)(aSrc + k0 + 32 + ac + 8);
      rb0 = *(const uint4*)(bSrc + k0 + 32 + bc);
      rb1 = *(const uint4*)(bSrc + k0 + 32 + bc + 8);
      if (k0 + 64 < K) {
        __builtin_prefetch(aSrc + k0 + 64 + ac, 0, 1);
        __builtin_prefetch(bSrc + k0 + 64 + bc, 0, 1);
      }
    }
    v16bf a0 = load_frag_a(As, 48, rowOfs, 0, lane);
    v16bf a1 = load_frag_a(As, 48, rowOfs + 16, 0, lane);
#pragma unroll
    for (int fn = 0; fn < 4; ++fn) {
      v16bf b = load_frag_bt(Bs, 40, colOfs + fn * 16, 0, lane);
      acc[0][fn] = wmma_bf16(a0, b, acc[0][fn]);
      acc[1][fn] = wmma_bf16(a1, b, acc[1][fn]);
    }
  }
}

#define GEMM_PROLOGUE()                                              \
  __shared__ __attribute__((aligned(16))) u16 As[128 * 48];          \
  __shared__ __attribute__((aligned(16))) u16 Bs[128 * 40];          \
  v8f acc[2][4];                                                     \
  _Pragma("unroll") for (int i = 0; i < 2; ++i)                      \
    _Pragma("unroll") for (int j = 0; j < 4; ++j) acc[i][j] = vzero8(); \
  int rowBase = blockIdx.x * 128, colBase = blockIdx.y * 128;

#define GEMM_EPI_SETUP()                                             \
  int lane = threadIdx.x & 31, wave = threadIdx.x >> 5;              \
  int rOfs = rowBase + (wave >> 1) * 32;                             \
  int cOfs = colBase + (wave & 1) * 64;                              \
  int hi = (lane >> 4) * 8, ln = lane & 15;

// ---------------- QKV GEMM ----------------
__global__ __launch_bounds__(256) void gemm_qkv_kernel(const u16* __restrict__ Xw,
                                                       const u16* __restrict__ Wq,
                                                       const float* __restrict__ qb,
                                                       u16* __restrict__ Qb, u16* __restrict__ Kb,
                                                       u16* __restrict__ Vb) {
  GEMM_PROLOGUE();
  gemm_core(Xw, Wq, DIM, DIM, rowBase, colBase, As, Bs, acc);
  GEMM_EPI_SETUP();
#pragma unroll
  for (int fm = 0; fm < 2; ++fm)
#pragma unroll
    for (int fn = 0; fn < 4; ++fn)
#pragma unroll
      for (int i = 0; i < 8; ++i) {
        int r = rOfs + fm * 16 + i + hi;
        int c = cOfs + fn * 16 + ln;
        float val = acc[fm][fn][i] + qb[c];
        int t = c / DIM, cc = c % DIM;
        int head = cc >> 5, d = cc & 31;
        int wi = r / NWIN, n = r % NWIN;
        size_t idx = (((size_t)(wi * HEADS + head)) * NWIN + n) * 32 + d;
        if (t == 0)      Qb[idx] = f2bf(val * SCALE);
        else if (t == 1) Kb[idx] = f2bf(val);
        else             Vb[idx] = f2bf(val);
      }
}

// ---------------- windowed attention: one wave per (window, head) ----------------
__global__ __launch_bounds__(64) void attn_kernel(const u16* __restrict__ Qb,
                                                  const u16* __restrict__ Kb,
                                                  const u16* __restrict__ Vb,
                                                  const float* __restrict__ biasW,
                                                  u16* __restrict__ Ob) {
  __shared__ __attribute__((aligned(16))) u16 sQ[2][64 * 48];  // A: rows, ld 48
  __shared__ __attribute__((aligned(16))) u16 sK[2][64 * 40];  // Bt for S: K rows natural, ld 40
  __shared__ __attribute__((aligned(16))) u16 sV[2][32 * 72];  // Bt for PV: Vt[d][n], ld 72
  __shared__ __attribute__((aligned(16))) u16 sP[2][64 * 72];  // A for PV: ld 72
  int wave = threadIdx.x >> 5, lane = threadIdx.x & 31;
  int wh = blockIdx.x * 2 + wave;
  int wi = wh / HEADS, head = wh % HEADS;
  u16* Q = sQ[wave]; u16* Kn = sK[wave]; u16* Vt = sV[wave]; u16* P = sP[wave];
  size_t base = (size_t)(wi * HEADS + head) * NWIN * 32;

  // stage Q and K (row-major 49x32 -> padded 64 rows), uint4 granularity
  for (int e = lane; e < 64 * 4; e += 32) {
    int row = e >> 2, c8 = (e & 3) * 8;
    uint4 vq = {0, 0, 0, 0}, vk = {0, 0, 0, 0};
    if (row < NWIN) {
      vq = *(const uint4*)(Qb + base + row * 32 + c8);
      vk = *(const uint4*)(Kb + base + row * 32 + c8);
    }
    *(uint4*)(Q + row * 48 + c8)  = vq;
    *(uint4*)(Kn + row * 40 + c8) = vk;
  }
  // stage V transposed: zero (incl pad) then scatter
  for (int e = lane; e < 32 * 72 / 2; e += 32) ((u32*)Vt)[e] = 0;
  for (int e = lane; e < NWIN * 32; e += 32) {
    int n = e >> 5, d = e & 31;
    Vt[d * 72 + n] = Vb[base + e];
  }

  // S = Q * K^T (64x64, K=32): 16 WMMAs
  v16bf af[4];
#pragma unroll
  for (int mt = 0; mt < 4; ++mt) af[mt] = load_frag_a(Q, 48, mt * 16, 0, lane);
  v8f S[4][4];
#pragma unroll
  for (int nt = 0; nt < 4; ++nt) {
    v16bf bfr = load_frag_bt(Kn, 40, nt * 16, 0, lane);
#pragma unroll
    for (int mt = 0; mt < 4; ++mt) S[mt][nt] = wmma_bf16(af[mt], bfr, vzero8());
  }

  // bias + mask + softmax; rows on this lane: mt*16 + i + 8*(lane>=16)
  const float* bh = biasW + head * (NWIN * NWIN);
  int hi = (lane >> 4) * 8, ln = lane & 15;
#pragma unroll
  for (int mt = 0; mt < 4; ++mt)
#pragma unroll
    for (int i = 0; i < 8; ++i) {
      int row = mt * 16 + i + hi;
      float sv[4];
#pragma unroll
      for (int nt = 0; nt < 4; ++nt) {
        int col = nt * 16 + ln;
        float s = S[mt][nt][i];
        if (col >= NWIN)      s = -1e30f;
        else if (row < NWIN)  s += bh[row * NWIN + col];
        sv[nt] = s;
      }
      float mx = fmaxf(fmaxf(sv[0], sv[1]), fmaxf(sv[2], sv[3]));
#pragma unroll
      for (int m = 1; m < 16; m <<= 1) mx = fmaxf(mx, __shfl_xor(mx, m, 32));
      float sum = 0.f;
#pragma unroll
      for (int nt = 0; nt < 4; ++nt) { float e = __expf(sv[nt] - mx); sv[nt] = e; sum += e; }
#pragma unroll
      for (int m = 1; m < 16; m <<= 1) sum += __shfl_xor(sum, m, 32);
      float inv = 1.f / sum;
#pragma unroll
      for (int nt = 0; nt < 4; ++nt) P[row * 72 + nt * 16 + ln] = f2bf(sv[nt] * inv);
    }

  // O = P * V (64x32, K=64): 16 WMMAs
  v8f O[4][2];
#pragma unroll
  for (int mt = 0; mt < 4; ++mt) { O[mt][0] = vzero8(); O[mt][1] = vzero8(); }
#pragma unroll
  for (int kk = 0; kk < 2; ++kk) {
    v16bf bv0 = load_frag_bt(Vt, 72, 0, kk * 32, lane);
    v16bf bv1 = load_frag_bt(Vt, 72, 16, kk * 32, lane);
#pragma unroll
    for (int mt = 0; mt < 4; ++mt) {
      v16bf ap = load_frag_a(P, 72, mt * 16, kk * 32, lane);
      O[mt][0] = wmma_bf16(ap, bv0, O[mt][0]);
      O[mt][1] = wmma_bf16(ap, bv1, O[mt][1]);
    }
  }
#pragma unroll
  for (int mt = 0; mt < 4; ++mt)
#pragma unroll
    for (int nd = 0; nd < 2; ++nd)
#pragma unroll
      for (int i = 0; i < 8; ++i) {
        int row = mt * 16 + i + hi;
        if (row < NWIN) {
          int d = nd * 16 + ln;
          Ob[((size_t)(wi * NWIN + row)) * DIM + head * 32 + d] = f2bf(O[mt][nd][i]);
        }
      }
}

// ---------------- proj GEMM + residual (window-major -> token-major) ----------------
__global__ __launch_bounds__(256) void gemm_proj_kernel(const u16* __restrict__ Ob,
                                                        const u16* __restrict__ Wp,
                                                        const float* __restrict__ pb,
                                                        const float* __restrict__ x,
                                                        float* __restrict__ x1) {
  GEMM_PROLOGUE();
  gemm_core(Ob, Wp, DIM, DIM, rowBase, colBase, As, Bs, acc);
  GEMM_EPI_SETUP();
#pragma unroll
  for (int fm = 0; fm < 2; ++fm)
#pragma unroll
    for (int fn = 0; fn < 4; ++fn)
#pragma unroll
      for (int i = 0; i < 8; ++i) {
        int r = rOfs + fm * 16 + i + hi;
        int c = cOfs + fn * 16 + ln;
        int t = row_to_token(r);
        x1[(size_t)t * DIM + c] = acc[fm][fn][i] + pb[c] + x[(size_t)t * DIM + c];
      }
}

// ---------------- MLP1 GEMM + exact GELU ----------------
__global__ __launch_bounds__(256) void gemm_mlp1_kernel(const u16* __restrict__ H2,
                                                        const u16* __restrict__ W1,
                                                        const float* __restrict__ b1,
                                                        u16* __restrict__ G) {
  GEMM_PROLOGUE();
  gemm_core(H2, W1, DIM, DIM, rowBase, colBase, As, Bs, acc);
  GEMM_EPI_SETUP();
#pragma unroll
  for (int fm = 0; fm < 2; ++fm)
#pragma unroll
    for (int fn = 0; fn < 4; ++fn)
#pragma unroll
      for (int i = 0; i < 8; ++i) {
        int r = rOfs + fm * 16 + i + hi;
        int c = cOfs + fn * 16 + ln;
        float v = acc[fm][fn][i] + b1[c];
        v = 0.5f * v * (1.f + erff(v * 0.70710678118654752f));
        G[(size_t)r * MLPH + c] = f2bf(v);
      }
}

// ---------------- MLP2 GEMM + residual -> output ----------------
__global__ __launch_bounds__(256) void gemm_mlp2_kernel(const u16* __restrict__ G,
                                                        const u16* __restrict__ W2,
                                                        const float* __restrict__ b2,
                                                        const float* __restrict__ x1,
                                                        float* __restrict__ out) {
  GEMM_PROLOGUE();
  gemm_core(G, W2, MLPH, MLPH, rowBase, colBase, As, Bs, acc);
  GEMM_EPI_SETUP();
#pragma unroll
  for (int fm = 0; fm < 2; ++fm)
#pragma unroll
    for (int fn = 0; fn < 4; ++fn)
#pragma unroll
      for (int i = 0; i < 8; ++i) {
        int r = rOfs + fm * 16 + i + hi;
        int c = cOfs + fn * 16 + ln;
        out[(size_t)r * DIM + c] = acc[fm][fn][i] + b2[c] + x1[(size_t)r * DIM + c];
      }
}

// ---------------- host-side orchestration ----------------
extern "C" void kernel_launch(void* const* d_in, const int* in_sizes, int n_in,
                              void* d_out, int out_size, void* d_ws, size_t ws_size,
                              hipStream_t stream) {
  const float* x      = (const float*)d_in[0];
  const float* ln1_g  = (const float*)d_in[1];
  const float* ln1_b  = (const float*)d_in[2];
  const float* qkv_w  = (const float*)d_in[3];
  const float* qkv_b  = (const float*)d_in[4];
  const float* rpb    = (const float*)d_in[5];
  const float* proj_w = (const float*)d_in[6];
  const float* proj_b = (const float*)d_in[7];
  const float* ln2_g  = (const float*)d_in[8];
  const float* ln2_b  = (const float*)d_in[9];
  const float* mlp_w1 = (const float*)d_in[10];
  const float* mlp_b1 = (const float*)d_in[11];
  const float* mlp_w2 = (const float*)d_in[12];
  const float* mlp_b2 = (const float*)d_in[13];
  const int*   ridx   = (const int*)d_in[14];
  float* out = (float*)d_out;

  char* ws = (char*)d_ws;
  size_t szXw  = (size_t)MTOT * DIM * 2;       // bf16 LN1 windows (later reused for Ob)
  size_t szQKV = (size_t)MTOT * 3 * DIM * 2;   // bf16 Q/K/V
  size_t offX1 = szXw + szQKV;                 // region A (reused for G)
  size_t offH2 = offX1 + (size_t)MTOT * DIM * 4;
  size_t offWq = offH2 + (size_t)MTOT * DIM * 2;
  size_t offWp = offWq + (size_t)DIM * 3 * DIM * 2;
  size_t offW1 = offWp + (size_t)DIM * DIM * 2;
  size_t offW2 = offW1 + (size_t)DIM * MLPH * 2;
  size_t offBi = offW2 + (size_t)MLPH * DIM * 2;

  u16*   Xw  = (u16*)(ws);
  u16*   Qb  = (u16*)(ws + szXw);
  u16*   Kb  = Qb + (size_t)MTOT * DIM;
  u16*   Vb  = Kb + (size_t)MTOT * DIM;
  u16*   Ob  = Xw;             // reuse: Xw dead after QKV GEMM
  u16*   G   = (u16*)(ws);     // reuse region A: exactly MTOT*1536*2 bytes
  float* x1  = (float*)(ws + offX1);
  u16*   H2  = (u16*)(ws + offH2);
  u16*   Wqb = (u16*)(ws + offWq);   // transposed [1152][384]
  u16*   Wpb = (u16*)(ws + offWp);   // transposed [384][384]
  u16*   W1b = (u16*)(ws + offW1);   // transposed [1536][384]
  u16*   W2b = (u16*)(ws + offW2);   // transposed [384][1536]
  float* biasW = (float*)(ws + offBi);

  // weight transpose+convert and bias expansion
  int nWq = DIM * 3 * DIM, nWp = DIM * DIM, nW1 = DIM * MLPH, nW2 = MLPH * DIM;
  transpose_bf16_kernel<<<(nWq + 255) / 256, 256, 0, stream>>>(qkv_w, Wqb, DIM, 3 * DIM);
  transpose_bf16_kernel<<<(nWp + 255) / 256, 256, 0, stream>>>(proj_w, Wpb, DIM, DIM);
  transpose_bf16_kernel<<<(nW1 + 255) / 256, 256, 0, stream>>>(mlp_w1, W1b, DIM, MLPH);
  transpose_bf16_kernel<<<(nW2 + 255) / 256, 256, 0, stream>>>(mlp_w2, W2b, MLPH, DIM);
  int nB = HEADS * NWIN * NWIN;
  bias_expand_kernel<<<(nB + 255) / 256, 256, 0, stream>>>(rpb, ridx, biasW);

  // LN1 + window partition -> bf16
  ln_kernel<<<MTOT / 8, 256, 0, stream>>>(x, ln1_g, ln1_b, Xw, 1);

  // QKV GEMM
  dim3 gq(MTOT / 128, (3 * DIM) / 128);
  gemm_qkv_kernel<<<gq, 256, 0, stream>>>(Xw, Wqb, qkv_b, Qb, Kb, Vb);

  // attention
  attn_kernel<<<NWTOT * HEADS / 2, 64, 0, stream>>>(Qb, Kb, Vb, biasW, Ob);

  // proj GEMM + residual (writes token-major x1)
  dim3 gp(MTOT / 128, DIM / 128);
  gemm_proj_kernel<<<gp, 256, 0, stream>>>(Ob, Wpb, proj_b, x, x1);

  // LN2
  ln_kernel<<<MTOT / 8, 256, 0, stream>>>(x1, ln2_g, ln2_b, H2, 0);

  // MLP
  dim3 g1(MTOT / 128, MLPH / 128);
  gemm_mlp1_kernel<<<g1, 256, 0, stream>>>(H2, W1b, mlp_b1, G);
  dim3 g2(MTOT / 128, DIM / 128);
  gemm_mlp2_kernel<<<g2, 256, 0, stream>>>(G, W2b, mlp_b2, x1, out);
}
